// MultiHeadAttention_51711406244477
// MI455X (gfx1250) — matrix operand order repaired and checked by reading, not compiled, not run on hardware
//
#include <hip/hip_runtime.h>
#include <stdint.h>

// ---------------------------------------------------------------- types
typedef __attribute__((ext_vector_type(16))) __bf16    v16bf;
typedef __attribute__((ext_vector_type(8)))  float     v8f;
typedef __attribute__((ext_vector_type(4)))  uint32_t  u32x4;

struct U32x8 { u32x4 lo, hi; };

static constexpr int BB = 2, DD = 512, NN = 2048, HH = 8, DKK = 64;

// ---------------------------------------------------------------- helpers
__device__ __forceinline__ uint16_t f2bf(float f) {
  uint32_t u = __builtin_bit_cast(uint32_t, f);
  u += 0x7FFFu + ((u >> 16) & 1u);           // round-to-nearest-even
  return (uint16_t)(u >> 16);
}

__device__ __forceinline__ v8f vzero() {
  v8f z;
#pragma unroll
  for (int j = 0; j < 8; ++j) z[j] = 0.0f;
  return z;
}

// Load a 16-element bf16 fragment as two 16B chunks at p and p+gap.
// A-fragment (16x32, row = lane&15): gap = 16, p = row*ld + kc + (lane>>4)*8
// B-fragment (32x16, col = lane&15): gap = 16, p = col*ld + kc + (lane>>4)*8
__device__ __forceinline__ v16bf make_frag(const uint16_t* p, int gap) {
  U32x8 u;
  u.lo = *(const u32x4*)p;
  u.hi = *(const u32x4*)(p + gap);
  return __builtin_bit_cast(v16bf, u);
}

__device__ __forceinline__ v8f wmma_bf16(v16bf a, v16bf b, v8f c) {
  return __builtin_amdgcn_wmma_f32_16x16x32_bf16(false, a, false, b,
                                                 (short)0, c, false, false);
}

// ---- CDNA5 async global -> LDS copy (ASYNCcnt-tracked), GV addressing.
// lds_byte_off: per-lane LDS byte address (HW adds wave LDS_BASE, ISA §10.2).
__device__ __forceinline__ void async_copy_b128(uint32_t lds_byte_off, const void* g) {
  asm volatile("global_load_async_to_lds_b128 %0, %1, off"
               :: "v"(lds_byte_off), "v"(g) : "memory");
}
__device__ __forceinline__ void wait_async0() {
  asm volatile("s_wait_asynccnt 0" ::: "memory");
}
__device__ __forceinline__ uint32_t lds_off(const void* p) {
  return (uint32_t)(uintptr_t)p;   // low 32 bits of a generic LDS pointer
}

// ---------------------------------------------------------------- converts
__global__ void convert_weights_kernel(const float* __restrict__ Wq,
                                       const float* __restrict__ Wk,
                                       const float* __restrict__ Wv,
                                       const float* __restrict__ Wm,
                                       const float* __restrict__ W1,
                                       const float* __restrict__ W2,
                                       uint16_t* __restrict__ wqkv,
                                       uint16_t* __restrict__ wm,
                                       uint16_t* __restrict__ w1,
                                       uint16_t* __restrict__ w2) {
  int i = blockIdx.x * blockDim.x + threadIdx.x;
  int stride = gridDim.x * blockDim.x;
  for (int t = i; t < 512 * 512; t += stride) {
    wqkv[t]          = f2bf(Wq[t]);
    wqkv[t + 262144] = f2bf(Wk[t]);
    wqkv[t + 524288] = f2bf(Wv[t]);
    wm[t]            = f2bf(Wm[t]);
  }
  for (int t = i; t < 1024 * 1024; t += stride) w1[t] = f2bf(W1[t]);
  for (int t = i; t < 512 * 1024; t += stride)  w2[t] = f2bf(W2[t]);
}

// transpose (B,D,N) f32 -> [b][n][c] bf16 layouts
__global__ void convert_acts_kernel(const float* __restrict__ xq,
                                    const float* __restrict__ xk,
                                    const float* __restrict__ xv,
                                    uint16_t* __restrict__ cat_t,   // [b][n][1024], writes c=512..1023
                                    uint16_t* __restrict__ xk_t,    // [b][n][512]
                                    uint16_t* __restrict__ xv_t) {  // [b][n][512]
  int i = blockIdx.x * blockDim.x + threadIdx.x;
  int stride = gridDim.x * blockDim.x;
  const int total = BB * DD * NN;
  for (int t = i; t < total; t += stride) {
    int b = t / (DD * NN);
    int r = t - b * DD * NN;
    int c = r / NN;
    int n = r - c * NN;
    size_t row = (size_t)b * NN + n;
    cat_t[row * 1024 + 512 + c] = f2bf(xq[t]);
    xk_t[row * 512 + c]         = f2bf(xk[t]);
    xv_t[row * 512 + c]         = f2bf(xv[t]);
  }
}

// ---------------------------------------------------------------- GEMM
// C[m][n] = sum_k A[m][k] * Bt[n][k] + bias[m]  (bf16 in, f32 acc)
// block tile 128(M) x 128(N), K-step 32; 8 waves, wave tile 32x64.
// Double-buffered LDS tiles filled by GLOBAL_LOAD_ASYNC_TO_LDS_B128.
struct GemmArgs {
  const uint16_t* A;
  const uint16_t* B0; const uint16_t* B1; const uint16_t* B2;
  int ldb0, ldb12;
  size_t bstr0, bstr12;
  int K;
  const float *bias0, *bias1, *bias2;
  uint16_t *o_q, *o_k, *o_v;     // QKV targets
  uint16_t *o_bf;                // merge / FF1 target ([n][c] bf16)
  size_t o_bstr;
  int o_ld;
  float *o_f32;                  // FF2 target (B,D,N)
  const float *rmean, *rvar, *gamma, *beta;
};

template <int MODE>
__global__ __launch_bounds__(256) void gemm_bf16_kernel(GemmArgs ga) {
  const int b  = blockIdx.z;
  const int n0 = blockIdx.x * 128;
  const int m0 = blockIdx.y * 128;

  const int tid  = threadIdx.x;
  const int wave = tid >> 5;
  const int lane = tid & 31;
  const int lrow = lane & 15;
  const int hf   = (lane >> 4) & 1;
  const int wm_  = wave >> 1;      // 0..3 (M)
  const int wn_  = wave & 1;       // 0..1 (N)

  const uint16_t* Bt; int ldb; const float* bias;
  int seg = 0;
  if constexpr (MODE == 0) {
    seg = blockIdx.y >> 2;
    if (seg == 0)      { Bt = ga.B0 + (size_t)b * ga.bstr0;  ldb = ga.ldb0;  bias = ga.bias0; }
    else if (seg == 1) { Bt = ga.B1 + (size_t)b * ga.bstr12; ldb = ga.ldb12; bias = ga.bias1; }
    else               { Bt = ga.B2 + (size_t)b * ga.bstr12; ldb = ga.ldb12; bias = ga.bias2; }
  } else {
    Bt = ga.B0 + (size_t)b * ga.bstr0; ldb = ga.ldb0; bias = ga.bias0;
  }

  __shared__ __align__(16) uint16_t sA[2][128 * 40];
  __shared__ __align__(16) uint16_t sB[2][128 * 40];

  v8f acc[2][4];
#pragma unroll
  for (int s = 0; s < 2; ++s)
#pragma unroll
    for (int t = 0; t < 4; ++t) acc[s][t] = vzero();

  const int arow = tid >> 1;
  const int acol = (tid & 1) * 16;

  // per-thread 32B of A and 32B of B per tile, issued as async b128 pairs
  auto issue_tiles = [&](int bufi, int k0) {
    const uint16_t* gA = &ga.A[(size_t)(m0 + arow) * ga.K + k0 + acol];
    const uint16_t* gB = &Bt[(size_t)(n0 + arow) * ldb + k0 + acol];
    uint32_t la = lds_off(&sA[bufi][arow * 40 + acol]);
    uint32_t lb = lds_off(&sB[bufi][arow * 40 + acol]);
    async_copy_b128(la,      gA);
    async_copy_b128(la + 16, gA + 8);
    async_copy_b128(lb,      gB);
    async_copy_b128(lb + 16, gB + 8);
  };

  issue_tiles(0, 0);
  wait_async0();
  __syncthreads();

  int cur = 0;
  for (int k0 = 0; k0 < ga.K; k0 += 32) {
    if (k0 + 32 < ga.K) issue_tiles(cur ^ 1, k0 + 32);   // prefetch next tile

    v16bf fa[2], fb[4];
#pragma unroll
    for (int s = 0; s < 2; ++s)
      fa[s] = make_frag(&sA[cur][(wm_ * 32 + s * 16 + lrow) * 40 + hf * 8], 16);
#pragma unroll
    for (int t = 0; t < 4; ++t)
      fb[t] = make_frag(&sB[cur][(wn_ * 64 + t * 16 + lrow) * 40 + hf * 8], 16);
#pragma unroll
    for (int s = 0; s < 2; ++s)
#pragma unroll
      for (int t = 0; t < 4; ++t)
        acc[s][t] = wmma_bf16(fa[s], fb[t], acc[s][t]);

    wait_async0();
    __syncthreads();
    cur ^= 1;
  }

  // ------------------------------------------------------------- epilogue
#pragma unroll
  for (int s = 0; s < 2; ++s) {
#pragma unroll
    for (int t = 0; t < 4; ++t) {
      const int col = n0 + wn_ * 64 + t * 16 + lrow;     // n
      const int mb  = m0 + wm_ * 32 + s * 16 + hf * 8;   // first of 8 m rows
      v8f a = acc[s][t];

      if constexpr (MODE == 0) {
        const int msegb = mb - seg * 512;
#pragma unroll
        for (int j = 0; j < 8; ++j) {
          int m  = msegb + j;
          int h  = m & 7;
          int dk = m >> 3;
          uint16_t bf = f2bf(a[j] + bias[m]);
          if (seg == 0)
            ga.o_q[(((size_t)b * HH + h) * NN + col) * DKK + dk] = bf;    // q[h][n][dk]
          else if (seg == 1)
            ga.o_k[(((size_t)b * HH + h) * NN + col) * DKK + dk] = bf;    // k[h][m][dk]
          else
            ga.o_v[(((size_t)b * HH + h) * DKK + dk) * NN + col] = bf;    // v[h][dk][m]
        }
      } else if constexpr (MODE == 1) {
        uint16_t tmp[8];
#pragma unroll
        for (int j = 0; j < 8; ++j) tmp[j] = f2bf(a[j] + bias[mb + j]);
        u32x4 pk;
#pragma unroll
        for (int p = 0; p < 4; ++p)
          pk[p] = (uint32_t)tmp[2 * p] | ((uint32_t)tmp[2 * p + 1] << 16);
        *(u32x4*)&ga.o_bf[(size_t)b * ga.o_bstr + (size_t)col * ga.o_ld + mb] = pk;
      } else if constexpr (MODE == 2) {
        uint16_t tmp[8];
#pragma unroll
        for (int j = 0; j < 8; ++j) {
          int m = mb + j;
          float v = a[j] + bias[m];
          float sc = ga.gamma[m] * __frsqrt_rn(ga.rvar[m] + 1e-5f);
          v = (v - ga.rmean[m]) * sc + ga.beta[m];
          v = fmaxf(v, 0.0f);
          tmp[j] = f2bf(v);
        }
        u32x4 pk;
#pragma unroll
        for (int p = 0; p < 4; ++p)
          pk[p] = (uint32_t)tmp[2 * p] | ((uint32_t)tmp[2 * p + 1] << 16);
        *(u32x4*)&ga.o_bf[(size_t)b * ga.o_bstr + (size_t)col * ga.o_ld + mb] = pk;
      } else {  // MODE == 3 : final f32 (B, D, N)
#pragma unroll
        for (int j = 0; j < 8; ++j)
          ga.o_f32[((size_t)b * DD + mb + j) * NN + col] = a[j] + bias[mb + j];
      }
    }
  }
}

// ---------------------------------------------------------------- attention
// per block: (b, h, 64 query rows). 4 waves, each owns 16 rows.
// q[h][n][dk], k[h][m][dk], v[h][dk][m]; out x_t[b][n][dk*H+h] bf16.
// K/V tiles double-buffered in LDS via async copies; all fragments hoisted
// ahead of the WMMA sequences so ds_loads pipeline under the matrix ops.
__global__ __launch_bounds__(128) void attn_kernel(const uint16_t* __restrict__ qh,
                                                   const uint16_t* __restrict__ kh,
                                                   const uint16_t* __restrict__ vt,
                                                   const int* __restrict__ mask,
                                                   uint16_t* __restrict__ xt) {
  const int b  = blockIdx.z;
  const int h  = blockIdx.y;
  const int q0 = blockIdx.x * 64;

  const int tid  = threadIdx.x;
  const int wave = tid >> 5;
  const int lane = tid & 31;
  const int lrow = lane & 15;
  const int hf   = (lane >> 4) & 1;

  __shared__ __align__(16) uint16_t sK[2][64 * 72];     // [m][dk]
  __shared__ __align__(16) uint16_t sV[2][64 * 72];     // [dk][m]
  __shared__ __align__(16) uint16_t sP[4][16 * 72];     // per-wave [row][m]

  const size_t head = ((size_t)b * HH + h) * (size_t)NN * DKK;
  const uint16_t* qb = qh + head;
  const uint16_t* kb = kh + head;
  const uint16_t* vb = vt + head;

  // Q fragments (A operand), held for the whole kernel
  const int qrow = q0 + wave * 16 + lrow;
  v16bf fq[2];
  fq[0] = make_frag(qb + (size_t)qrow * DKK + 0  + hf * 8, 16);
  fq[1] = make_frag(qb + (size_t)qrow * DKK + 32 + hf * 8, 16);

  v8f acc_o[4];
#pragma unroll
  for (int t = 0; t < 4; ++t) acc_o[t] = vzero();
  float run_m[8], run_l[8];
#pragma unroll
  for (int j = 0; j < 8; ++j) { run_m[j] = -1e30f; run_l[j] = 0.0f; }

  const int r  = tid >> 1;          // 0..63
  const int cb = (tid & 1) * 32;

  auto issue_kv = [&](int bufi, int m0) {
    const uint16_t* gK = &kb[(size_t)(m0 + r) * DKK + cb];
    const uint16_t* gV = &vb[(size_t)r * NN + m0 + cb];
    uint32_t lk = lds_off(&sK[bufi][r * 72 + cb]);
    uint32_t lv = lds_off(&sV[bufi][r * 72 + cb]);
#pragma unroll
    for (int c = 0; c < 4; ++c) {
      async_copy_b128(lk + c * 16, gK + c * 8);
      async_copy_b128(lv + c * 16, gV + c * 8);
    }
  };

  issue_kv(0, 0);
  wait_async0();
  __syncthreads();

  int cur = 0;
  for (int m0 = 0; m0 < NN; m0 += 64) {
    if (m0 + 64 < NN) issue_kv(cur ^ 1, m0 + 64);        // prefetch next KV tile

    // --- hoist all K fragments, then S = Q K^T (16 rows x 64 cols per wave)
    v16bf fk[8];
#pragma unroll
    for (int t = 0; t < 4; ++t) {
      fk[2 * t]     = make_frag(&sK[cur][(t * 16 + lrow) * 72 + 0  + hf * 8], 16);
      fk[2 * t + 1] = make_frag(&sK[cur][(t * 16 + lrow) * 72 + 32 + hf * 8], 16);
    }
    v8f sc[4];
#pragma unroll
    for (int t = 0; t < 4; ++t) {
      sc[t] = wmma_bf16(fq[0], fk[2 * t], vzero());
      sc[t] = wmma_bf16(fq[1], fk[2 * t + 1], sc[t]);
    }

    // --- scale + mask
#pragma unroll
    for (int t = 0; t < 4; ++t) {
      const int colm = m0 + t * 16 + lrow;
#pragma unroll
      for (int j = 0; j < 8; ++j) {
        const int n = q0 + wave * 16 + hf * 8 + j;
        float v = sc[t][j] * 0.125f;                     // 1/sqrt(64)
        if (mask[((size_t)b * NN + n) * NN + colm] == 0) v = -1e9f;
        sc[t][j] = v;
      }
    }

    // --- online softmax (row stats per lane: rows hf*8+j live on this half)
#pragma unroll
    for (int j = 0; j < 8; ++j) {
      float mx = fmaxf(fmaxf(sc[0][j], sc[1][j]), fmaxf(sc[2][j], sc[3][j]));
      mx = fmaxf(mx, __shfl_xor(mx, 1));
      mx = fmaxf(mx, __shfl_xor(mx, 2));
      mx = fmaxf(mx, __shfl_xor(mx, 4));
      mx = fmaxf(mx, __shfl_xor(mx, 8));
      const float mnew = fmaxf(run_m[j], mx);
      const float fs   = __expf(run_m[j] - mnew);
      run_m[j] = mnew;
      float rs = 0.0f;
#pragma unroll
      for (int t = 0; t < 4; ++t) {
        float p = __expf(sc[t][j] - mnew);
        sc[t][j] = p;
        rs += p;
      }
      rs += __shfl_xor(rs, 1);
      rs += __shfl_xor(rs, 2);
      rs += __shfl_xor(rs, 4);
      rs += __shfl_xor(rs, 8);
      run_l[j] = run_l[j] * fs + rs;
#pragma unroll
      for (int t = 0; t < 4; ++t) acc_o[t][j] *= fs;
    }

    // --- stage P (C layout -> A layout via LDS)
#pragma unroll
    for (int t = 0; t < 4; ++t)
#pragma unroll
      for (int j = 0; j < 8; ++j)
        sP[wave][(hf * 8 + j) * 72 + t * 16 + lrow] = f2bf(sc[t][j]);

    // --- O += P V ; P fragments are dkt-invariant (2 total), V hoisted (8)
    v16bf fp[2];
    fp[0] = make_frag(&sP[wave][lrow * 72 + 0  + hf * 8], 16);
    fp[1] = make_frag(&sP[wave][lrow * 72 + 32 + hf * 8], 16);
    v16bf fv[8];
#pragma unroll
    for (int dkt = 0; dkt < 4; ++dkt) {
      fv[2 * dkt]     = make_frag(&sV[cur][(dkt * 16 + lrow) * 72 + 0  + hf * 8], 16);
      fv[2 * dkt + 1] = make_frag(&sV[cur][(dkt * 16 + lrow) * 72 + 32 + hf * 8], 16);
    }
#pragma unroll
    for (int dkt = 0; dkt < 4; ++dkt) {
      acc_o[dkt] = wmma_bf16(fp[0], fv[2 * dkt], acc_o[dkt]);
      acc_o[dkt] = wmma_bf16(fp[1], fv[2 * dkt + 1], acc_o[dkt]);
    }

    wait_async0();
    __syncthreads();
    cur ^= 1;
  }

  // --- normalize + store x_t[b][n][dk*H + h]
#pragma unroll
  for (int dkt = 0; dkt < 4; ++dkt) {
#pragma unroll
    for (int j = 0; j < 8; ++j) {
      const int n  = q0 + wave * 16 + hf * 8 + j;
      const int dk = dkt * 16 + lrow;
      const float o = acc_o[dkt][j] / run_l[j];
      xt[((size_t)b * NN + n) * DD + dk * HH + h] = f2bf(o);
    }
  }
}

// ---------------------------------------------------------------- launch
extern "C" void kernel_launch(void* const* d_in, const int* in_sizes, int n_in,
                              void* d_out, int out_size, void* d_ws, size_t ws_size,
                              hipStream_t stream) {
  const float* init_query = (const float*)d_in[0];
  const float* key_t      = (const float*)d_in[1];
  const float* value      = (const float*)d_in[2];
  const int*   mask       = (const int*)d_in[3];
  const float* Wq = (const float*)d_in[4];  const float* bq = (const float*)d_in[5];
  const float* Wk = (const float*)d_in[6];  const float* bk = (const float*)d_in[7];
  const float* Wv = (const float*)d_in[8];  const float* bv = (const float*)d_in[9];
  const float* Wm = (const float*)d_in[10]; const float* bm = (const float*)d_in[11];
  const float* W1 = (const float*)d_in[12]; const float* b1 = (const float*)d_in[13];
  const float* gamma = (const float*)d_in[14]; const float* beta = (const float*)d_in[15];
  const float* rmean = (const float*)d_in[16]; const float* rvar = (const float*)d_in[17];
  const float* W2 = (const float*)d_in[18]; const float* b2 = (const float*)d_in[19];

  uint8_t* ws = (uint8_t*)d_ws;
  size_t off = 0;
  auto alloc16 = [&](size_t elems) -> uint16_t* {
    uint16_t* p = (uint16_t*)(ws + off);
    off += (elems * 2 + 255) & ~(size_t)255;
    return p;
  };
  uint16_t* cat_t = alloc16((size_t)BB * NN * 1024);      // [b][n][1024]
  uint16_t* xk_t  = alloc16((size_t)BB * NN * 512);
  uint16_t* xv_t  = alloc16((size_t)BB * NN * 512);
  uint16_t* q_h   = alloc16((size_t)BB * HH * NN * DKK);
  uint16_t* k_h   = alloc16((size_t)BB * HH * NN * DKK);
  uint16_t* v_t   = alloc16((size_t)BB * HH * DKK * NN);
  uint16_t* x_t   = alloc16((size_t)BB * NN * 512);
  uint16_t* h_t   = alloc16((size_t)BB * NN * 1024);
  uint16_t* wqkv  = alloc16((size_t)1536 * 512);
  uint16_t* wm_b  = alloc16((size_t)512 * 512);
  uint16_t* w1_b  = alloc16((size_t)1024 * 1024);
  uint16_t* w2_b  = alloc16((size_t)512 * 1024);

  convert_weights_kernel<<<512, 256, 0, stream>>>(Wq, Wk, Wv, Wm, W1, W2,
                                                  wqkv, wm_b, w1_b, w2_b);
  convert_acts_kernel<<<1024, 256, 0, stream>>>(init_query, key_t, value,
                                                cat_t, xk_t, xv_t);

  // QKV projections (stacked M = 1536)
  {
    GemmArgs g{};
    g.A = wqkv;
    g.B0 = cat_t + 512; g.ldb0 = 1024; g.bstr0 = (size_t)NN * 1024;
    g.B1 = xk_t; g.B2 = xv_t; g.ldb12 = 512; g.bstr12 = (size_t)NN * 512;
    g.K = 512;
    g.bias0 = bq; g.bias1 = bk; g.bias2 = bv;
    g.o_q = q_h; g.o_k = k_h; g.o_v = v_t;
    gemm_bf16_kernel<0><<<dim3(16, 12, 2), 256, 0, stream>>>(g);
  }

  attn_kernel<<<dim3(32, 8, 2), 128, 0, stream>>>(q_h, k_h, v_t, mask, x_t);

  // merge conv -> cat_t[:, :, 0:512]
  {
    GemmArgs g{};
    g.A = wm_b;
    g.B0 = x_t; g.ldb0 = 512; g.bstr0 = (size_t)NN * 512;
    g.K = 512; g.bias0 = bm;
    g.o_bf = cat_t; g.o_bstr = (size_t)NN * 1024; g.o_ld = 1024;
    gemm_bf16_kernel<1><<<dim3(16, 4, 2), 256, 0, stream>>>(g);
  }

  // FF1 + BN + ReLU -> h_t
  {
    GemmArgs g{};
    g.A = w1_b;
    g.B0 = cat_t; g.ldb0 = 1024; g.bstr0 = (size_t)NN * 1024;
    g.K = 1024; g.bias0 = b1;
    g.rmean = rmean; g.rvar = rvar; g.gamma = gamma; g.beta = beta;
    g.o_bf = h_t; g.o_bstr = (size_t)NN * 1024; g.o_ld = 1024;
    gemm_bf16_kernel<2><<<dim3(16, 8, 2), 256, 0, stream>>>(g);
  }

  // FF2 -> d_out (B, D, N) f32
  {
    GemmArgs g{};
    g.A = w2_b;
    g.B0 = h_t; g.ldb0 = 1024; g.bstr0 = (size_t)NN * 1024;
    g.K = 1024; g.bias0 = b2;
    g.o_f32 = (float*)d_out;
    gemm_bf16_kernel<3><<<dim3(16, 4, 2), 256, 0, stream>>>(g);
  }
}
